// Block_72395968741402
// MI455X (gfx1250) — matrix operand order repaired and checked by reading, not compiled, not run on hardware
//
#include <hip/hip_runtime.h>
#include <math.h>

typedef _Float16 f16;
typedef __attribute__((ext_vector_type(16))) _Float16 v16h;
typedef __attribute__((ext_vector_type(8)))  _Float16 v8h;
typedef __attribute__((ext_vector_type(8)))  float    v8f;

#define CAT16(lo, hi) __builtin_shufflevector((lo), (hi), 0,1,2,3,4,5,6,7,8,9,10,11,12,13,14,15)

static __device__ __forceinline__ v8f wmma_f16(v16h a, v16h b, v8f c) {
  // D = A(16x32 f16) * B(32x16 f16) + C(16x16 f32)
  return __builtin_amdgcn_wmma_f32_16x16x32_f16(false, a, false, b, (short)0, c, false, false);
}

static __device__ __forceinline__ v8f v8f_zero() {
  v8f z = {0.f,0.f,0.f,0.f,0.f,0.f,0.f,0.f};
  return z;
}

// ---- DPP 16-lane xor-butterfly reduction (no DS, no dscnt waits) ----------
// quad_perm(1,0,3,2)=0xB1, quad_perm(2,3,0,1)=0x4E,
// row_half_mirror=0x141 (combines quads), row_mirror=0x140 (combines octets).
template<int CTRL>
static __device__ __forceinline__ float dpp_movf(float v) {
  return __int_as_float(
      __builtin_amdgcn_update_dpp(0, __float_as_int(v), CTRL, 0xf, 0xf, true));
}
static __device__ __forceinline__ float red_max16(float v) {
  v = fmaxf(v, dpp_movf<0xB1>(v));
  v = fmaxf(v, dpp_movf<0x4E>(v));
  v = fmaxf(v, dpp_movf<0x141>(v));
  v = fmaxf(v, dpp_movf<0x140>(v));
  return v;
}

static __device__ __forceinline__ float gelu_f(float t) {
  const float c = 0.79788456080286535588f; // sqrt(2/pi)
  return 0.5f * t * (1.0f + tanhf(c * (t + 0.044715f * t * t * t)));
}

// ---------------------------------------------------------------- converts
__global__ void cvt_f32_f16(const float* __restrict__ src, f16* __restrict__ dst, int n) {
  int i = blockIdx.x * blockDim.x + threadIdx.x;
  int stride = gridDim.x * blockDim.x;
  for (; i < n; i += stride) dst[i] = (f16)src[i];
}

// ---------------------------------------------------------------- layernorm -> f16
__global__ __launch_bounds__(256)
void layernorm_f16(const float* __restrict__ x, const float* __restrict__ g,
                   const float* __restrict__ beta, f16* __restrict__ out, int D) {
  const int row = blockIdx.x;
  const int t = threadIdx.x;               // 256 threads, D = 1024 -> float4 each
  const float4 v = ((const float4*)(x + (size_t)row * D))[t];
  float s  = v.x + v.y + v.z + v.w;
  float s2 = v.x*v.x + v.y*v.y + v.z*v.z + v.w*v.w;
  #pragma unroll
  for (int off = 16; off > 0; off >>= 1) {
    s  += __shfl_xor(s,  off, 32);
    s2 += __shfl_xor(s2, off, 32);
  }
  __shared__ float ps[8], ps2[8];
  if ((t & 31) == 0) { ps[t >> 5] = s; ps2[t >> 5] = s2; }
  __syncthreads();
  s = 0.f; s2 = 0.f;
  #pragma unroll
  for (int i = 0; i < 8; ++i) { s += ps[i]; s2 += ps2[i]; }
  const float mu   = s / (float)D;
  const float rstd = rsqrtf(s2 / (float)D - mu * mu + 1e-5f);
  const float4 gv = ((const float4*)g)[t];
  const float4 bv = ((const float4*)beta)[t];
  f16* o = out + (size_t)row * D + t * 4;
  o[0] = (f16)((v.x - mu) * rstd * gv.x + bv.x);
  o[1] = (f16)((v.y - mu) * rstd * gv.y + bv.y);
  o[2] = (f16)((v.z - mu) * rstd * gv.z + bv.z);
  o[3] = (f16)((v.w - mu) * rstd * gv.w + bv.w);
}

// ---------------------------------------------------------------- WMMA GEMM
// C[M,N] = A[M,K] @ W[N,K]^T  (+ fused epilogue per MODE)
// MODE 1: QKV   -> q16/k16 [B,H,S,64] f16 (q pre-scaled 1/8), v transposed [B,H,64,S]
// MODE 2: outF = residual + acc + bias           (attn out-proj / residual)
// MODE 3: outH = f16(gelu(acc + bias))           (fc1)
// MODE 4: outF = residual + acc + bias           (fc2 -> d_out)
template<int MODE>
__global__ __launch_bounds__(128)
void gemm_wmma_f16(const f16* __restrict__ A, const f16* __restrict__ W,
                   int M, int N, int K,
                   const float* __restrict__ bias,
                   const float* __restrict__ residual,
                   float* __restrict__ outF,
                   f16* __restrict__ outH,
                   const float* __restrict__ bq, const float* __restrict__ bk,
                   const float* __restrict__ bv,
                   f16* __restrict__ q16, f16* __restrict__ k16, f16* __restrict__ vt16,
                   int S, int H) {
  const int lane = threadIdx.x & 31;
  const int wave = threadIdx.x >> 5;
  const int lm   = lane & 15;       // M-row / N-col within a 16-tile
  const int hf   = lane >> 4;       // half-wave select
  const int n0 = blockIdx.x * 64;
  const int m0 = blockIdx.y * 128 + wave * 32;

  v8f acc[2][4];
  #pragma unroll
  for (int mt = 0; mt < 2; ++mt)
    #pragma unroll
    for (int nt = 0; nt < 4; ++nt) acc[mt][nt] = v8f_zero();

  for (int k0 = 0; k0 < K; k0 += 32) {
    v16h afr[2];
    #pragma unroll
    for (int mt = 0; mt < 2; ++mt) {
      // 16-bit A layout: per lane, halves 0..7 at K = k0 + hf*8, halves 8..15 at +16
      const f16* p = A + (size_t)(m0 + mt * 16 + lm) * K + k0 + hf * 8;
      v8h lo = *(const v8h*)p;
      v8h hi = *(const v8h*)(p + 16);
      afr[mt] = CAT16(lo, hi);
    }
    v16h bfr[4];
    #pragma unroll
    for (int nt = 0; nt < 4; ++nt) {
      // 16-bit B layout: per lane, halves 0..7 at K = k0 + hf*8, halves 8..15 at +16
      const f16* p = W + (size_t)(n0 + nt * 16 + lm) * K + k0 + hf * 8;
      v8h lo = *(const v8h*)p;
      v8h hi = *(const v8h*)(p + 16);
      bfr[nt] = CAT16(lo, hi);
    }
    #pragma unroll
    for (int mt = 0; mt < 2; ++mt)
      #pragma unroll
      for (int nt = 0; nt < 4; ++nt)
        acc[mt][nt] = wmma_f16(afr[mt], bfr[nt], acc[mt][nt]);
  }

  #pragma unroll
  for (int mt = 0; mt < 2; ++mt) {
    #pragma unroll
    for (int nt = 0; nt < 4; ++nt) {
      #pragma unroll
      for (int r = 0; r < 8; ++r) {
        const int m = m0 + mt * 16 + r + hf * 8;   // C layout: row = r + 8*half
        const int n = n0 + nt * 16 + lm;           //           col = lane & 15
        const float v = acc[mt][nt][r];
        if constexpr (MODE == 1) {
          const int b_ = m / S, s = m % S;
          const int which = n >> 10, hh = (n & 1023) >> 6, e = n & 63;
          const int bh = b_ * H + hh;
          if (which == 0)
            q16[((size_t)bh * S + s) * 64 + e] = (f16)((v + bq[hh * 64 + e]) * 0.125f);
          else if (which == 1)
            k16[((size_t)bh * S + s) * 64 + e] = (f16)(v + bk[hh * 64 + e]);
          else
            vt16[((size_t)bh * 64 + e) * S + s] = (f16)(v + bv[hh * 64 + e]);
        } else if constexpr (MODE == 3) {
          outH[(size_t)m * N + n] = (f16)gelu_f(v + bias[n]);
        } else { // MODE 2 / 4: residual + bias
          const size_t idx = (size_t)m * N + n;
          outF[idx] = residual[idx] + v + bias[n];
        }
      }
    }
  }
}

// ---------------------------------------------------------------- flash attention
// grid = (S/64, B*H), block = 128 (4 waves, 16 query rows each)
__global__ __launch_bounds__(128)
void attention_wmma(const f16* __restrict__ q16, const f16* __restrict__ k16,
                    const f16* __restrict__ vt16, f16* __restrict__ attn16,
                    int S, int H) {
  const int lane = threadIdx.x & 31;
  const int wave = threadIdx.x >> 5;
  const int lm = lane & 15;
  const int hf = lane >> 4;
  const int bh = blockIdx.y;
  const int b_ = bh / H, h = bh % H;
  const int q0 = blockIdx.x * 64 + wave * 16;
  const int Dfull = H * 64;

  const f16* Q  = q16  + (size_t)bh * S * 64;
  const f16* Kp = k16  + (size_t)bh * S * 64;
  const f16* Vt = vt16 + (size_t)bh * 64 * S;

  __shared__ __align__(16) f16 pbuf[4][16][64];   // per-wave private P staging

  // all-ones B fragment: rowsum(P) = P @ ones(64x16) on the matrix pipe
  v16h ones;
  #pragma unroll
  for (int j = 0; j < 16; ++j) ones[j] = (f16)1.0f;

  // Q fragments for K-dim (head dim 64 -> two 32-wide steps); q is pre-scaled.
  v16h qa[2];
  #pragma unroll
  for (int ks = 0; ks < 2; ++ks) {
    const f16* p = Q + (size_t)(q0 + lm) * 64 + ks * 32 + hf * 8;
    v8h lo = *(const v8h*)p;
    v8h hi = *(const v8h*)(p + 16);
    qa[ks] = CAT16(lo, hi);
  }

  float mi[8], li[8];
  #pragma unroll
  for (int r = 0; r < 8; ++r) { mi[r] = -1e30f; li[r] = 0.f; }
  v8f o[4];
  #pragma unroll
  for (int nt = 0; nt < 4; ++nt) o[nt] = v8f_zero();

  for (int kt = 0; kt < S; kt += 64) {
    // ---- scores: 16 queries x 64 keys, 8 WMMAs
    v8f sc[4];
    #pragma unroll
    for (int nt = 0; nt < 4; ++nt) sc[nt] = v8f_zero();
    #pragma unroll
    for (int ks = 0; ks < 2; ++ks) {
      #pragma unroll
      for (int nt = 0; nt < 4; ++nt) {
        const f16* p = Kp + (size_t)(kt + nt * 16 + lm) * 64 + ks * 32 + hf * 8;
        v8h lo = *(const v8h*)p;
        v8h hi = *(const v8h*)(p + 16);
        sc[nt] = wmma_f16(qa[ks], CAT16(lo, hi), sc[nt]);
      }
    }
    // ---- online softmax max (row = r + 8*hf, col = lane&15 + 16*nt), DPP only
    float mnew[8], corr[8];
    #pragma unroll
    for (int r = 0; r < 8; ++r) {
      float v = fmaxf(fmaxf(sc[0][r], sc[1][r]), fmaxf(sc[2][r], sc[3][r]));
      v = red_max16(v);
      mnew[r] = fmaxf(mi[r], v);
      corr[r] = __expf(mi[r] - mnew[r]);
      mi[r] = mnew[r];
    }
    #pragma unroll
    for (int nt = 0; nt < 4; ++nt)
      #pragma unroll
      for (int r = 0; r < 8; ++r)
        sc[nt][r] = __expf(sc[nt][r] - mnew[r]);
    #pragma unroll
    for (int nt = 0; nt < 4; ++nt)
      #pragma unroll
      for (int r = 0; r < 8; ++r) o[nt][r] *= corr[r];
    // ---- stage P into LDS to re-shape C layout -> A layout
    #pragma unroll
    for (int nt = 0; nt < 4; ++nt)
      #pragma unroll
      for (int r = 0; r < 8; ++r)
        pbuf[wave][r + hf * 8][nt * 16 + lm] = (f16)sc[nt][r];
    // ---- P @ V, plus rowsum(P) via ones-matrix WMMA (keys are the K-dim)
    v8f rsum = v8f_zero();
    #pragma unroll
    for (int ks = 0; ks < 2; ++ks) {
      const f16* pp = &pbuf[wave][lm][ks * 32 + hf * 8];
      v8h plo = *(const v8h*)pp;
      v8h phi = *(const v8h*)(pp + 16);
      v16h pa = CAT16(plo, phi);
      rsum = wmma_f16(pa, ones, rsum);
      #pragma unroll
      for (int nt = 0; nt < 4; ++nt) {
        const f16* vp = Vt + (size_t)(nt * 16 + lm) * S + kt + ks * 32 + hf * 8;
        v8h lo = *(const v8h*)vp;
        v8h hi = *(const v8h*)(vp + 16);
        o[nt] = wmma_f16(pa, CAT16(lo, hi), o[nt]);
      }
    }
    #pragma unroll
    for (int r = 0; r < 8; ++r) li[r] = li[r] * corr[r] + rsum[r];
  }

  // ---- finalize: O /= l, write f16 [B,S,D] for the out-projection GEMM
  #pragma unroll
  for (int r = 0; r < 8; ++r) {
    const float inv = 1.f / li[r];
    const int srow = q0 + r + hf * 8;
    #pragma unroll
    for (int nt = 0; nt < 4; ++nt) {
      const int e = h * 64 + nt * 16 + lm;
      attn16[((size_t)b_ * S + srow) * Dfull + e] = (f16)(o[nt][r] * inv);
    }
  }
}

// ---------------------------------------------------------------- launcher
extern "C" void kernel_launch(void* const* d_in, const int* in_sizes, int n_in,
                              void* d_out, int out_size, void* d_ws, size_t ws_size,
                              hipStream_t stream) {
  (void)in_sizes; (void)n_in; (void)out_size; (void)ws_size;
  constexpr int B = 2, S = 2048, D = 1024, H = 16, F = 4096;
  constexpr int M = B * S;

  const float* x   = (const float*)d_in[0];
  const float* Wq  = (const float*)d_in[1];
  const float* bq  = (const float*)d_in[2];
  const float* Wk  = (const float*)d_in[3];
  const float* bk  = (const float*)d_in[4];
  const float* Wv  = (const float*)d_in[5];
  const float* bv  = (const float*)d_in[6];
  const float* Wo  = (const float*)d_in[7];
  const float* bo  = (const float*)d_in[8];
  const float* g1  = (const float*)d_in[9];
  const float* be1 = (const float*)d_in[10];
  const float* W1  = (const float*)d_in[11];
  const float* b1  = (const float*)d_in[12];
  const float* W2  = (const float*)d_in[13];
  const float* b2  = (const float*)d_in[14];
  const float* g2  = (const float*)d_in[15];
  const float* be2 = (const float*)d_in[16];

  char* p = (char*)d_ws;
  auto carve = [&](size_t bytes) -> void* {
    void* r = (void*)p;
    p += (bytes + 255) & ~(size_t)255;
    return r;
  };
  f16*   h16    = (f16*)carve((size_t)M * D * 2);
  f16*   wqkv16 = (f16*)carve((size_t)3 * D * D * 2);
  f16*   wo16   = (f16*)carve((size_t)D * D * 2);
  f16*   w1_16  = (f16*)carve((size_t)F * D * 2);
  f16*   w2_16  = (f16*)carve((size_t)D * F * 2);
  f16*   q16    = (f16*)carve((size_t)M * D * 2);
  f16*   k16    = (f16*)carve((size_t)M * D * 2);
  f16*   vt16   = (f16*)carve((size_t)M * D * 2);
  f16*   attn16 = (f16*)carve((size_t)M * D * 2);
  f16*   act16  = (f16*)carve((size_t)M * F * 2);
  float* x2     = (float*)carve((size_t)M * D * 4);

  // weights -> f16 (QKV packed row-wise: [Wq; Wk; Wv] each [1024,1024])
  cvt_f32_f16<<<2048, 256, 0, stream>>>(Wq, wqkv16,             D * D);
  cvt_f32_f16<<<2048, 256, 0, stream>>>(Wk, wqkv16 + D * D,     D * D);
  cvt_f32_f16<<<2048, 256, 0, stream>>>(Wv, wqkv16 + 2 * D * D, D * D);
  cvt_f32_f16<<<2048, 256, 0, stream>>>(Wo, wo16,  D * D);
  cvt_f32_f16<<<2048, 256, 0, stream>>>(W1, w1_16, F * D);
  cvt_f32_f16<<<2048, 256, 0, stream>>>(W2, w2_16, D * F);

  // LN1
  layernorm_f16<<<M, 256, 0, stream>>>(x, g1, be1, h16, D);
  // QKV projection (fused bias, q-scale, V-transpose)
  gemm_wmma_f16<1><<<dim3(3 * D / 64, M / 128), 128, 0, stream>>>(
      h16, wqkv16, M, 3 * D, D,
      nullptr, nullptr, nullptr, nullptr,
      bq, bk, bv, q16, k16, vt16, S, H);
  // flash attention
  attention_wmma<<<dim3(S / 64, B * H), 128, 0, stream>>>(q16, k16, vt16, attn16, S, H);
  // out-projection + residual
  gemm_wmma_f16<2><<<dim3(D / 64, M / 128), 128, 0, stream>>>(
      attn16, wo16, M, D, D,
      bo, x, x2, nullptr,
      nullptr, nullptr, nullptr, nullptr, nullptr, nullptr, S, H);
  // LN2
  layernorm_f16<<<M, 256, 0, stream>>>(x2, g2, be2, h16, D);
  // fc1 + GELU
  gemm_wmma_f16<3><<<dim3(F / 64, M / 128), 128, 0, stream>>>(
      h16, w1_16, M, F, D,
      b1, nullptr, nullptr, act16,
      nullptr, nullptr, nullptr, nullptr, nullptr, nullptr, S, H);
  // fc2 + residual -> output
  gemm_wmma_f16<4><<<dim3(D / 64, M / 128), 128, 0, stream>>>(
      act16, w2_16, M, D, F,
      b2, x2, (float*)d_out, nullptr,
      nullptr, nullptr, nullptr, nullptr, nullptr, nullptr, S, H);
}
